// LGDSR_rgb_32057635897519
// MI455X (gfx1250) — compile-verified, matrix operand order checked by reading, not run on hardware
//
#include <hip/hip_runtime.h>
#include <hip/hip_bf16.h>

typedef __attribute__((ext_vector_type(16))) _Float16 v16h;
typedef __attribute__((ext_vector_type(8)))  float    v8f;

#define OMEGA 30.0f
// OMEGA / (2*pi): feeds v_sin_f32 directly (it takes revolutions), folding the
// SIREN frequency and the radians->revolutions conversion into ONE multiply.
#define KSIN 4.774648292756860f
#define HSRC 360
#define WSRC 640
#define HDST 540
#define WDST 960
#define NPIX (2 * HDST * WDST)          // 1,036,800
#define NTILES (NPIX / 16)              // 64,800
#define WAVES_PER_BLOCK 8
#define NFRAG 50                        // 32 (L2) + 16 (L3) + 2 (L4)

// sin(OMEGA * z) via the hardware transcendental pipe (single v_sin_f32).
__device__ __forceinline__ float sin_omega(float z) {
    return __builtin_amdgcn_sinf(z * KSIN);
}

// Maps (vgpr index v, lane-half, odd) -> local K within a 16x32 f16 A/B fragment
// per CDNA5 ISA 7.12.2 "16-bit A-Matrix 16x32" layout.
__device__ __forceinline__ int kloc(int v, int halfK, int odd) {
    int base = (v < 4) ? (2 * v) : (16 + 2 * (v - 4));
    return base + 8 * halfK + odd;
}

__device__ __forceinline__ float sample_border(const float* __restrict__ img,
                                               float gx, float gy) {
    // matches grid_sample(bilinear, border, align_corners=False)
    float x = ((gx + 1.0f) * (float)WSRC - 1.0f) * 0.5f;
    float y = ((gy + 1.0f) * (float)HSRC - 1.0f) * 0.5f;
    float x0f = floorf(x), y0f = floorf(y);
    float wx = x - x0f, wy = y - y0f;
    int x0 = (int)x0f, y0 = (int)y0f;
    int x0c = min(max(x0, 0), WSRC - 1);
    int x1c = min(max(x0 + 1, 0), WSRC - 1);
    int y0c = min(max(y0, 0), HSRC - 1);
    int y1c = min(max(y0 + 1, 0), HSRC - 1);
    float v00 = img[y0c * WSRC + x0c], v01 = img[y0c * WSRC + x1c];
    float v10 = img[y1c * WSRC + x0c], v11 = img[y1c * WSRC + x1c];
    float top = v00 + (v01 - v00) * wx;
    float bot = v10 + (v11 - v10) * wx;
    return top + (bot - top) * wy;
}

// ---------------------------------------------------------------------------
// Prep: convert f32 weights into f16 WMMA B-fragments (pre-swizzled) in d_ws.
// frag 0..31  : w2 [128x128]  (nt 0..7, kt 0..3)
// frag 32..47 : w3 [128x64]   (nt 0..3, kt 0..3)
// frag 48..49 : w4 [64x15->16](nt 0,    kt 0..1)  (n=15 zero padded)
// Element (frag, lane, j): B[k][n] with k = 32*kt + kloc(j/2, lane/16, j%2),
// n = 16*nt + lane%16. Stored at wfrag[(frag*32 + lane)*16 + j].
// ---------------------------------------------------------------------------
__global__ void lgdsr_prep_weights(const float* __restrict__ w2,
                                   const float* __restrict__ w3,
                                   const float* __restrict__ w4,
                                   _Float16* __restrict__ wfrag) {
    int idx = blockIdx.x * blockDim.x + threadIdx.x;
    if (idx >= NFRAG * 512) return;
    int frag = idx >> 9;
    int r    = idx & 511;
    int lane = r >> 4;
    int j    = r & 15;
    int v = j >> 1, odd = j & 1, halfK = lane >> 4, nl = lane & 15;

    const float* w; int fin, fout, nt, kt;
    if (frag < 32)      { w = w2; fin = 128; fout = 128; nt = frag >> 2;        kt = frag & 3; }
    else if (frag < 48) { w = w3; fin = 128; fout = 64;  nt = (frag - 32) >> 2; kt = (frag - 32) & 3; }
    else                { w = w4; fin = 64;  fout = 15;  nt = 0;                kt = frag - 48; }

    int k = 32 * kt + kloc(v, halfK, odd);
    int n = 16 * nt + nl;
    float val = (k < fin && n < fout) ? w[k * fout + n] : 0.0f;
    wfrag[idx] = (_Float16)val;
}

// ---------------------------------------------------------------------------
// Main: one wave32 per 16-pixel tile. SIREN MLP via v_wmma_f32_16x16x32_f16.
// ---------------------------------------------------------------------------
__global__ __launch_bounds__(WAVES_PER_BLOCK * 32)
void lgdsr_main(const float* __restrict__ lr,
                const float* __restrict__ w1, const float* __restrict__ b1,
                const float* __restrict__ b2, const float* __restrict__ b3,
                const float* __restrict__ b4,
                const _Float16* __restrict__ wfrag,
                float* __restrict__ out) {
    __shared__ _Float16 lds_h[WAVES_PER_BLOCK][16 * 128];  // 32 KB

    const int tid   = threadIdx.x;
    const int wave  = tid >> 5;
    const int lane  = tid & 31;
    const int m     = lane & 15;     // pixel row within tile / matrix M
    const int halfK = lane >> 4;     // which K-half of the A fragment

    // ----- pixel coordinates -----
    const int T  = blockIdx.x * WAVES_PER_BLOCK + wave;   // tile id (exact cover)
    const int p  = T * 16 + m;
    const int bi = p / (HDST * WDST);
    const int rem = p - bi * (HDST * WDST);
    const int oy = rem / WDST;
    const int ox = rem - oy * WDST;

    const float gx = ((float)ox + 0.5f) * (2.0f / (float)WDST) - 1.0f;
    const float gy = ((float)oy + 0.5f) * (2.0f / (float)HDST) - 1.0f;

    // ----- edge features (green channel) -----
    const float* green = lr + ((size_t)bi * 3 + 1) * (HSRC * WSRC);
    const float pixg = sample_border(green, gx, gy);

    const int sx = (int)floorf(((float)ox + 0.5f) * (1.0f / 1.5f) - 0.5f);
    const int sy = (int)floorf(((float)oy + 0.5f) * (1.0f / 1.5f) - 0.5f);
    auto tap = [&](int dx, int dy) -> float {
        int xi = min(max(sx + dx, 0), WSRC - 1);
        int yi = min(max(sy + dy, 0), HSRC - 1);
        return green[yi * WSRC + xi];
    };
    const float tf = tap(0, 0), tg = tap(1, 0), tj = tap(0, 1), tk = tap(1, 1);
    const float edge_vote = fabsf(tf - tj) + fabsf(pixg - tj) + fabsf(pixg - tf);
    const bool  maskv = edge_vote > (8.0f / 255.0f);
    const float dxv = tk + tg - (tf + tj);
    const float dyv = tk + tj - (tf + tg);
    const float slope = atanf(dyv / (dxv + 1e-5f)) * (1.0f / 3.1415926f) + 0.5f;
    const float strength = sqrtf(dxv * dxv + dyv * dyv);

    // ----- layer 1 (2 -> 128): compute straight into A-fragment layout -----
    v16h a2[4];
    #pragma unroll
    for (int kt = 0; kt < 4; ++kt) {
        #pragma unroll
        for (int v = 0; v < 8; ++v) {
            #pragma unroll
            for (int odd = 0; odd < 2; ++odd) {
                int f = 32 * kt + kloc(v, halfK, odd);
                float h = sin_omega(slope * w1[f] + strength * w1[128 + f] + b1[f]);
                a2[kt][2 * v + odd] = (_Float16)h;
            }
        }
    }

    _Float16* hbuf = &lds_h[wave][0];

    // ----- layer 2 (128 -> 128): 8 N-tiles x 4 K-steps of WMMA -----
    #pragma unroll
    for (int nt = 0; nt < 8; ++nt) {
        v8f acc = {};
        #pragma unroll
        for (int kt = 0; kt < 4; ++kt) {
            const v16h bf = *(const v16h*)(wfrag + ((size_t)((nt * 4 + kt) * 32 + lane)) * 16);
            acc = __builtin_amdgcn_wmma_f32_16x16x32_f16(false, a2[kt], false, bf,
                                                         (short)0, acc, false, false);
        }
        int n = nt * 16 + m;
        float bn = b2[n];
        #pragma unroll
        for (int r = 0; r < 8; ++r)
            hbuf[(r + 8 * halfK) * 128 + n] = (_Float16)sin_omega(acc[r] + bn);
    }
    asm volatile("s_wait_dscnt 0" ::: "memory");

    // re-transpose: gather layer-3 A fragments from LDS
    v16h a3[4];
    #pragma unroll
    for (int kt = 0; kt < 4; ++kt)
        #pragma unroll
        for (int v = 0; v < 8; ++v)
            #pragma unroll
            for (int odd = 0; odd < 2; ++odd)
                a3[kt][2 * v + odd] = hbuf[m * 128 + 32 * kt + kloc(v, halfK, odd)];
    asm volatile("s_wait_dscnt 0" ::: "memory");

    // ----- layer 3 (128 -> 64): 4 N-tiles x 4 K-steps -----
    #pragma unroll
    for (int nt = 0; nt < 4; ++nt) {
        v8f acc = {};
        #pragma unroll
        for (int kt = 0; kt < 4; ++kt) {
            const v16h bf = *(const v16h*)(wfrag + ((size_t)((32 + nt * 4 + kt) * 32 + lane)) * 16);
            acc = __builtin_amdgcn_wmma_f32_16x16x32_f16(false, a3[kt], false, bf,
                                                         (short)0, acc, false, false);
        }
        int n = nt * 16 + m;
        float bn = b3[n];
        #pragma unroll
        for (int r = 0; r < 8; ++r)
            hbuf[(r + 8 * halfK) * 128 + n] = (_Float16)sin_omega(acc[r] + bn);
    }
    asm volatile("s_wait_dscnt 0" ::: "memory");

    v16h a4[2];
    #pragma unroll
    for (int kt = 0; kt < 2; ++kt)
        #pragma unroll
        for (int v = 0; v < 8; ++v)
            #pragma unroll
            for (int odd = 0; odd < 2; ++odd)
                a4[kt][2 * v + odd] = hbuf[m * 128 + 32 * kt + kloc(v, halfK, odd)];
    asm volatile("s_wait_dscnt 0" ::: "memory");

    // ----- layer 4 (64 -> 15, N padded to 16): 2 K-steps -----
    v8f acc4 = {};
    #pragma unroll
    for (int kt = 0; kt < 2; ++kt) {
        const v16h bf = *(const v16h*)(wfrag + ((size_t)((48 + kt) * 32 + lane)) * 16);
        acc4 = __builtin_amdgcn_wmma_f32_16x16x32_f16(false, a4[kt], false, bf,
                                                      (short)0, acc4, false, false);
    }
    // stage mlp_out [16 pixels][16 outs] in f32 (reuse wave's LDS region)
    float* mf = (float*)hbuf;
    {
        int n = m;
        float bn = (n < 15) ? b4[n] : 0.0f;
        #pragma unroll
        for (int r = 0; r < 8; ++r)
            mf[(r + 8 * halfK) * 16 + n] = acc4[r] + bn;
    }
    asm volatile("s_wait_dscnt 0" ::: "memory");

    // ----- final: 5-tap learned resample + mask select (lanes 0..15) -----
    if (lane < 16) {
        const float DX[5] = {0.f, -1.f, 0.f, 1.f, 0.f};
        const float DY[5] = {-1.f, 0.f, 0.f, 0.f, 1.f};
        const float WI[5] = {0.15f, 0.15f, 0.4f, 0.15f, 0.15f};
        float wgt[5], gxs[5], gys[5], wsum = 0.0f;
        #pragma unroll
        for (int s = 0; s < 5; ++s) {
            float offx = mf[m * 16 + s * 2 + 0] + DX[s];
            float offy = mf[m * 16 + s * 2 + 1] + DY[s];
            gxs[s] = gx + offx * (2.0f / (float)WDST);
            gys[s] = gy + offy * (2.0f / (float)HDST);
            wgt[s] = mf[m * 16 + 10 + s] + WI[s];
            wsum += wgt[s];
        }
        const float inv_wsum = 1.0f / wsum;
        #pragma unroll
        for (int c = 0; c < 3; ++c) {
            const float* img = lr + ((size_t)bi * 3 + c) * (HSRC * WSRC);
            float accc = 0.0f;
            #pragma unroll
            for (int s = 0; s < 5; ++s)
                accc += sample_border(img, gxs[s], gys[s]) * wgt[s];
            float enh  = accc * inv_wsum;
            float pixc = sample_border(img, gx, gy);
            out[(((size_t)bi * 3 + c) * HDST + oy) * WDST + ox] = maskv ? enh : pixc;
        }
    }
}

extern "C" void kernel_launch(void* const* d_in, const int* in_sizes, int n_in,
                              void* d_out, int out_size, void* d_ws, size_t ws_size,
                              hipStream_t stream) {
    const float* lr = (const float*)d_in[0];
    const float* w1 = (const float*)d_in[1];
    const float* b1 = (const float*)d_in[2];
    const float* w2 = (const float*)d_in[3];
    const float* b2 = (const float*)d_in[4];
    const float* w3 = (const float*)d_in[5];
    const float* b3 = (const float*)d_in[6];
    const float* w4 = (const float*)d_in[7];
    const float* b4 = (const float*)d_in[8];
    float* out = (float*)d_out;
    _Float16* wfrag = (_Float16*)d_ws;   // needs 50*512*2 = 51,200 bytes

    // 1) weights -> pre-swizzled f16 WMMA fragments
    {
        int total = NFRAG * 512;
        int blk = 256;
        lgdsr_prep_weights<<<(total + blk - 1) / blk, blk, 0, stream>>>(w2, w3, w4, wfrag);
    }
    // 2) main fused kernel: grid exactly covers NTILES with 8 waves/block
    {
        int blocks = NTILES / WAVES_PER_BLOCK;   // 8100
        lgdsr_main<<<blocks, WAVES_PER_BLOCK * 32, 0, stream>>>(
            lr, w1, b1, b2, b3, b4, wfrag, out);
    }
}
